// DecisionTransformerGPT2Attention_8581344657773
// MI455X (gfx1250) — matrix operand order repaired and checked by reading, not compiled, not run on hardware
//
#include <hip/hip_runtime.h>
#include <hip/hip_bf16.h>

typedef __attribute__((ext_vector_type(16))) __bf16       v16bf;
typedef __attribute__((ext_vector_type(8)))  float        v8f;
typedef __attribute__((ext_vector_type(4)))  unsigned int v4u;
typedef __attribute__((ext_vector_type(8)))  int          v8i;
typedef __attribute__((ext_vector_type(4)))  int          v4i;

// Problem constants: B=2, S=2048, D=1024, H=16, HD=64
#define SEQ   2048
#define DIM   1024
#define QKVN  3072
#define HEADS 16

// ---------- bf16 helpers ----------
__device__ inline __bf16 to_bf(float f) {
  unsigned u = __builtin_bit_cast(unsigned, f);
  u = (u + 0x7FFFu + ((u >> 16) & 1u)) >> 16;
  return __builtin_bit_cast(__bf16, (unsigned short)u);
}
__device__ inline unsigned pack_bf2(float x, float y) {
  return (unsigned)__builtin_bit_cast(unsigned short, to_bf(x)) |
         ((unsigned)__builtin_bit_cast(unsigned short, to_bf(y)) << 16);
}
__device__ inline __bf16 bf_lo(unsigned u) { return __builtin_bit_cast(__bf16, (unsigned short)(u & 0xFFFFu)); }
__device__ inline __bf16 bf_hi(unsigned u) { return __builtin_bit_cast(__bf16, (unsigned short)(u >> 16)); }

// ---------- TDM: 2D tile -> LDS (6-arg builtin; D# per ISA 08 §8) ----------
// dsize_code: 0=1B, 1=2B, 2=4B; rowlen/row_stride/tensor_rows in elements.
__device__ inline void tdm_load_2d(unsigned lds_off, const void* gptr,
                                   unsigned rows, unsigned rowlen,
                                   unsigned row_stride, unsigned tensor_rows,
                                   unsigned dsize_code) {
  unsigned long long ga = (unsigned long long)(size_t)gptr;
  v4u g0;
  g0[0] = 1u;                                   // [1:0] count=1
  g0[1] = lds_off;                              // [63:32] lds_addr (bytes)
  g0[2] = (unsigned)(ga & 0xFFFFFFFFu);         // [95:64] global_addr lo
  g0[3] = (unsigned)((ga >> 32) & 0x01FFFFFFu)  // [120:96] global_addr hi
        | (2u << 30);                           // [127:126] type=2 (image)
  v8i g1;
  g1[0] = (int)(dsize_code << 16);                                    // data_size; wg_mask=0
  g1[1] = (int)((rowlen & 0xFFFFu) << 16);                            // tensor_dim0 lo16
  g1[2] = (int)(((rowlen >> 16) & 0xFFFFu) | ((tensor_rows & 0xFFFFu) << 16)); // dim0 hi / dim1 lo
  g1[3] = (int)(((tensor_rows >> 16) & 0xFFFFu) | ((rowlen & 0xFFFFu) << 16)); // dim1 hi / tile_dim0
  g1[4] = (int)(rows & 0xFFFFu);                                      // tile_dim1 (tile_dim2=0)
  g1[5] = (int)row_stride;                                            // tensor_dim0_stride lo32
  g1[6] = 0;
  g1[7] = 0;
  v4i z4 = {0, 0, 0, 0};
  v8i z8 = {0, 0, 0, 0, 0, 0, 0, 0};
  __builtin_amdgcn_tensor_load_to_lds(g0, g1, z4, z4, z8, 0);
}

// ---------- WMMA fragment loaders (CDNA5 16-bit A 16x32 / B 32x16 layouts) ----------
// A fragment, f32 source, contiguous K: float2 per packed pair (converts).
__device__ inline v16bf a_frag_f32c(const float* __restrict__ base, size_t strideM) {
  const int lane = threadIdx.x & 31;
  const int half = lane >> 4;
  const float* row = base + (size_t)(lane & 15) * strideM;
  v16bf a;
#pragma unroll
  for (int v = 0; v < 8; ++v) {
    const int k = (v & 3) * 2 + half * 8 + (v >> 2) * 16;
    float2 f = *(const float2*)(row + k);
    a[2 * v]     = to_bf(f.x);
    a[2 * v + 1] = to_bf(f.y);
  }
  return a;
}

// A fragment, bf16 source, contiguous K: one b32 per packed pair, no converts.
__device__ inline v16bf a_frag_bf16c(const __bf16* __restrict__ base, size_t strideM) {
  const int lane = threadIdx.x & 31;
  const int half = lane >> 4;
  const __bf16* row = base + (size_t)(lane & 15) * strideM;
  v16bf a;
#pragma unroll
  for (int v = 0; v < 8; ++v) {
    const int k = (v & 3) * 2 + half * 8 + (v >> 2) * 16;
    const unsigned u = *(const unsigned*)(row + k);
    a[2 * v]     = bf_lo(u);
    a[2 * v + 1] = bf_hi(u);
  }
  return a;
}

// B fragment, bf16 source, contiguous K (element (k,n) = base[n*strideN + k]), no converts.
__device__ inline v16bf b_frag_bf16c(const __bf16* __restrict__ base, size_t strideN) {
  const int lane = threadIdx.x & 31;
  const __bf16* col = base + (size_t)(lane & 15) * strideN + (lane >> 4) * 8;
  v16bf b;
#pragma unroll
  for (int v = 0; v < 8; ++v) {
    const unsigned u = *(const unsigned*)(col + (v & 3) * 2 + (v >> 2) * 16);
    b[2 * v]     = bf_lo(u);
    b[2 * v + 1] = bf_hi(u);
  }
  return b;
}

__device__ inline v8f wmma_bf16(v16bf a, v16bf b, v8f c) {
  return __builtin_amdgcn_wmma_f32_16x16x32_bf16(false, a, false, b, (short)0, c, false, false);
}

// ===================== Kernel 0: weights -> transposed bf16 (one-time) =====================
__global__ void __launch_bounds__(256) k_prep(const float* __restrict__ Wa,
                                              const float* __restrict__ Wp,
                                              __bf16* __restrict__ WaT,
                                              __bf16* __restrict__ WpT) {
  const size_t i = (size_t)blockIdx.x * 256 + threadIdx.x;
  if (i < (size_t)QKVN * DIM) {
    const int n = (int)(i / DIM), k = (int)(i % DIM);
    WaT[i] = to_bf(Wa[(size_t)k * QKVN + n]);
  }
  if (i < (size_t)DIM * DIM) {
    const int n = (int)(i / DIM), k = (int)(i % DIM);
    WpT[i] = to_bf(Wp[(size_t)k * DIM + n]);
  }
}

// ===================== Kernel 1: QKV = X @ W_attn + b (bf16 out) =====================
// TDM stages 16x1024 f32 X-slab; block converts it once to a bf16 slab in LDS.
// 8 waves x 4 N-tiles; grid: 256 M-tiles x 6 N-groups = 1536 blocks.
__global__ void __launch_bounds__(256) k_qkv(const float* __restrict__ X,
                                             const __bf16* __restrict__ WaT,
                                             const float* __restrict__ bias,
                                             __bf16* __restrict__ qkv) {
  __shared__ __align__(16) unsigned char ldsbuf[16 * DIM * 4 + 16 * DIM * 2];
  float*  Aslab = (float*)ldsbuf;                     // TDM destination (LDS offset 0)
  __bf16* Abf   = (__bf16*)(ldsbuf + 16 * DIM * 4);
  const int tM = blockIdx.x / 6;
  const int ng = blockIdx.x % 6;
  if (threadIdx.x < 32) {
    tdm_load_2d(0u, X + (size_t)tM * 16 * DIM, 16u, DIM, DIM, 4096u, 2u);
    __builtin_amdgcn_s_wait_tensorcnt(0);
  }
  __syncthreads();
  {
    const float2* src = (const float2*)Aslab;
    unsigned*     dst = (unsigned*)Abf;
    for (int i = threadIdx.x; i < 16 * DIM / 2; i += 256)
      dst[i] = pack_bf2(src[i].x, src[i].y);
  }
  __syncthreads();
  const int tN0 = ng * 32 + (threadIdx.x >> 5) * 4;
  v8f c[4] = {};
  for (int k0 = 0; k0 < DIM; k0 += 32) {
    v16bf a = a_frag_bf16c(Abf + k0, DIM);            // 2x ds_load_b128
#pragma unroll
    for (int j = 0; j < 4; ++j) {
      v16bf b = b_frag_bf16c(WaT + (size_t)(tN0 + j) * 16 * DIM + k0, DIM);
      c[j] = wmma_bf16(a, b, c[j]);
    }
  }
  const int lane = threadIdx.x & 31;
  const int n  = lane & 15;
  const int mh = (lane >> 4) * 8;
#pragma unroll
  for (int j = 0; j < 4; ++j) {
    const float bv = bias[(tN0 + j) * 16 + n];
    __bf16* out = qkv + (size_t)tM * 16 * QKVN + (size_t)(tN0 + j) * 16 + n;
#pragma unroll
    for (int v = 0; v < 8; ++v) out[(size_t)(v + mh) * QKVN] = to_bf(c[j][v] + bv);
  }
}

// ===================== Kernel 1b: V -> per-head transposed bf16 VT[b][h][d][s] =====================
// Coalesced writes along s; one-time 8 MB pass. Makes PV B-fragments contiguous-K.
__global__ void __launch_bounds__(256) k_vt(const __bf16* __restrict__ qkv,
                                            __bf16* __restrict__ VT) {
  const size_t i = (size_t)blockIdx.x * 256 + threadIdx.x;   // [bh][d][s] flat
  if (i >= (size_t)2 * HEADS * 64 * SEQ) return;
  const int s  = (int)(i & (SEQ - 1));
  const int d  = (int)((i >> 11) & 63);
  const int bh = (int)(i >> 17);
  const int b = bh >> 4, h = bh & 15;
  VT[i] = qkv[((size_t)b * SEQ + s) * QKVN + 2 * DIM + h * 64 + d];
}

// ===================== Kernel 2: scores = QK^T / 8 with causal mask =====================
// Wave: one q-tile x 4 key-tiles; Q fragments loaded once, reused 4x.
// Grid: 32 bh * 128 qt * 32 ktg = 131072 waves -> 16384 blocks.
__global__ void __launch_bounds__(256) k_scores(const __bf16* __restrict__ qkv,
                                                float* __restrict__ Wts) {
  const int wid = blockIdx.x * 8 + (threadIdx.x >> 5);
  const int bh  = wid >> 12;
  const int t   = wid & 4095;
  const int qt  = t >> 5;
  const int kt0 = (t & 31) * 4;
  const int b = bh >> 4, h = bh & 15;
  const int lane = threadIdx.x & 31;
  const int n  = lane & 15;
  const int mh = (lane >> 4) * 8;
  const float MASKV = -3.402823466e38f;  // finfo(f32).min
  float* orow = Wts + ((size_t)bh * SEQ + (size_t)qt * 16) * SEQ;
  if (kt0 > qt) {
#pragma unroll
    for (int j = 0; j < 4; ++j) {
      float* out = orow + (size_t)(kt0 + j) * 16 + n;
#pragma unroll
      for (int v = 0; v < 8; ++v) out[(size_t)(v + mh) * SEQ] = MASKV;
    }
    return;
  }
  const __bf16* qb = qkv + ((size_t)b * SEQ + (size_t)qt * 16) * QKVN + h * 64;
  const v16bf a0 = a_frag_bf16c(qb, QKVN);
  const v16bf a1 = a_frag_bf16c(qb + 32, QKVN);
#pragma unroll
  for (int j = 0; j < 4; ++j) {
    const int kt = kt0 + j;
    float* out = orow + (size_t)kt * 16 + n;
    if (kt > qt) {
#pragma unroll
      for (int v = 0; v < 8; ++v) out[(size_t)(v + mh) * SEQ] = MASKV;
      continue;
    }
    const __bf16* kb = qkv + ((size_t)b * SEQ + (size_t)kt * 16) * QKVN + DIM + h * 64;
    v8f c = {};
    c = wmma_bf16(a0, b_frag_bf16c(kb, QKVN), c);     // K^T: k = head-dim (contig), n = key
    c = wmma_bf16(a1, b_frag_bf16c(kb + 32, QKVN), c);
#pragma unroll
    for (int v = 0; v < 8; ++v) {
      const int gq = qt * 16 + v + mh;
      const int gk = kt * 16 + n;
      out[(size_t)(v + mh) * SEQ] = (gk <= gq) ? c[v] * 0.125f : MASKV;
    }
  }
}

// ===================== Kernel 3: row softmax (in-place, f32) =====================
__global__ void __launch_bounds__(256) k_softmax(float* __restrict__ W) {
  __shared__ float red[256];
  float* p = W + (size_t)blockIdx.x * SEQ;
  const int tid = threadIdx.x;
  float m = -3.402823466e38f;
  for (int i = tid; i < SEQ; i += 256) m = fmaxf(m, p[i]);
  red[tid] = m; __syncthreads();
  for (int s = 128; s > 0; s >>= 1) {
    if (tid < s) red[tid] = fmaxf(red[tid], red[tid + s]);
    __syncthreads();
  }
  m = red[0]; __syncthreads();
  float sum = 0.0f;
  for (int i = tid; i < SEQ; i += 256) {
    float e = __expf(p[i] - m);   // masked -> exp(-inf) = 0
    p[i] = e;
    sum += e;
  }
  red[tid] = sum; __syncthreads();
  for (int s = 128; s > 0; s >>= 1) {
    if (tid < s) red[tid] += red[tid + s];
    __syncthreads();
  }
  const float inv = 1.0f / red[0];
  for (int i = tid; i < SEQ; i += 256) p[i] *= inv;
}

// ===================== Kernel 4: attn_out = P @ V, bf16 out (causal-bounded K loop) =====================
// Wave: one q-tile, all 4 head-dim tiles; V read via VT (contiguous-K packed pairs).
// Grid: 32 bh * 128 qt = 4096 waves -> 512 blocks.
__global__ void __launch_bounds__(256) k_av(const float* __restrict__ P,
                                            const __bf16* __restrict__ VT,
                                            __bf16* __restrict__ AO) {
  const int wid = blockIdx.x * 8 + (threadIdx.x >> 5);
  const int bh = wid >> 7;
  const int qt = wid & 127;
  const int b = bh >> 4, h = bh & 15;
  const float* prow = P + ((size_t)bh * SEQ + (size_t)qt * 16) * SEQ;
  const __bf16* vt = VT + (size_t)bh * 64 * SEQ;   // VT[bh][d][s]
  v8f c[4] = {};
  const int kend = qt * 16 + 16;  // later keys have exactly-zero probability
  for (int k0 = 0; k0 < kend; k0 += 32) {
    v16bf a = a_frag_f32c(prow + k0, SEQ);
#pragma unroll
    for (int j = 0; j < 4; ++j) {
      // element (k=key, n=d) = VT[bh][j*16+n][k0+key]
      v16bf bb = b_frag_bf16c(vt + (size_t)j * 16 * SEQ + k0, SEQ);
      c[j] = wmma_bf16(a, bb, c[j]);
    }
  }
  const int lane = threadIdx.x & 31;
  const int n  = lane & 15;
  const int mh = (lane >> 4) * 8;
#pragma unroll
  for (int j = 0; j < 4; ++j) {
    __bf16* out = AO + ((size_t)b * SEQ + (size_t)qt * 16) * DIM + h * 64 + j * 16 + n;
#pragma unroll
    for (int v = 0; v < 8; ++v) out[(size_t)(v + mh) * DIM] = to_bf(c[j][v]);
  }
}

// ===================== Kernel 5: Y = AO @ W_proj + b (f32 out) =====================
// TDM stages 16x1024 bf16 A-slab (data_size=2B): zero conversions in the loop.
// Grid: 256 M-tiles x 2 N-groups = 512 blocks.
__global__ void __launch_bounds__(256) k_proj(const __bf16* __restrict__ AO,
                                              const __bf16* __restrict__ WpT,
                                              const float* __restrict__ bias,
                                              float* __restrict__ Y) {
  __shared__ __align__(16) __bf16 Abf[16 * DIM];   // 32 KB, single shared object at LDS offset 0
  const int tM = blockIdx.x >> 1;
  const int ng = blockIdx.x & 1;
  if (threadIdx.x < 32) {
    tdm_load_2d(0u, AO + (size_t)tM * 16 * DIM, 16u, DIM, DIM, 4096u, 1u);
    __builtin_amdgcn_s_wait_tensorcnt(0);
  }
  __syncthreads();
  const int tN0 = ng * 32 + (threadIdx.x >> 5) * 4;
  v8f c[4] = {};
  for (int k0 = 0; k0 < DIM; k0 += 32) {
    v16bf a = a_frag_bf16c(Abf + k0, DIM);
#pragma unroll
    for (int j = 0; j < 4; ++j) {
      v16bf b = b_frag_bf16c(WpT + (size_t)(tN0 + j) * 16 * DIM + k0, DIM);
      c[j] = wmma_bf16(a, b, c[j]);
    }
  }
  const int lane = threadIdx.x & 31;
  const int n  = lane & 15;
  const int mh = (lane >> 4) * 8;
#pragma unroll
  for (int j = 0; j < 4; ++j) {
    const float bv = bias[(tN0 + j) * 16 + n];
    float* out = Y + (size_t)tM * 16 * DIM + (size_t)(tN0 + j) * 16 + n;
#pragma unroll
    for (int v = 0; v < 8; ++v) out[(size_t)(v + mh) * DIM] = c[j][v] + bv;
  }
}

extern "C" void kernel_launch(void* const* d_in, const int* in_sizes, int n_in,
                              void* d_out, int out_size, void* d_ws, size_t ws_size,
                              hipStream_t stream) {
  (void)in_sizes; (void)n_in; (void)out_size; (void)ws_size;
  const float* X     = (const float*)d_in[0];   // [2,2048,1024]
  const float* Wattn = (const float*)d_in[1];   // [1024,3072]
  const float* Battn = (const float*)d_in[2];   // [3072]
  const float* Wproj = (const float*)d_in[3];   // [1024,1024]
  const float* Bproj = (const float*)d_in[4];   // [1024]

  float* Y   = (float*)d_out;                   // attn_output [2,2048,1024]
  float* Wts = Y + (size_t)2 * SEQ * DIM;       // attn_weights [2,16,2048,2048]

  // workspace layout (bf16 intermediates)
  char* ws = (char*)d_ws;
  __bf16* qkv = (__bf16*)ws;  ws += (size_t)4096 * QKVN * 2;        // 24 MB
  __bf16* AO  = (__bf16*)ws;  ws += (size_t)4096 * DIM  * 2;        //  8 MB
  __bf16* WaT = (__bf16*)ws;  ws += (size_t)QKVN * DIM  * 2;        //  6 MB
  __bf16* WpT = (__bf16*)ws;  ws += (size_t)DIM  * DIM  * 2;        //  2 MB
  __bf16* VT  = (__bf16*)ws;                                        //  8 MB: [2][16][64][2048]

  k_prep   <<<12288, 256, 0, stream>>>(Wattn, Wproj, WaT, WpT);
  k_qkv    <<<1536,  256, 0, stream>>>(X, WaT, Battn, qkv);
  k_vt     <<<16384, 256, 0, stream>>>(qkv, VT);
  k_scores <<<16384, 256, 0, stream>>>(qkv, Wts);
  k_softmax<<<65536, 256, 0, stream>>>(Wts);
  k_av     <<<512,   256, 0, stream>>>(Wts, VT, AO);
  k_proj   <<<512,   256, 0, stream>>>(AO, WpT, Bproj, Y);
}
